// QValueNet_59854664237414
// MI455X (gfx1250) — matrix operand order repaired and checked by reading, not compiled, not run on hardware
//
#include <hip/hip_runtime.h>
#include <hip/hip_bf16.h>
#include <stdint.h>

// ---------------------------------------------------------------------------
// Problem constants (from the reference)
// ---------------------------------------------------------------------------
#define BATCH 4096
#define NJ    28
#define XF    126
#define AF    9
#define HDIM  512
#define MROWS (BATCH * NJ)      // 114688
#define K1PAD 160               // 137 padded to mult of 32
#define K23PAD (K1PAD + HDIM)   // 672
#define TM    64                // rows per workgroup in fused kernel

typedef __attribute__((ext_vector_type(16))) _Float16 v16h;
typedef __attribute__((ext_vector_type(8)))  _Float16 v8h;
typedef __attribute__((ext_vector_type(8)))  float    v8f;

// ---------------------------------------------------------------------------
// Fragment loaders (CDNA5 WMMA 16-bit layouts, wave32)
//
// A (16x32, MxK): lane 0-15 = row M=lane, holds K = kb..kb+7 (elems 0..7)
//                 and K = kb+16..kb+23 (elems 8..15), kb = k0 + (lane>>4)*8.
// B (32x16, KxN): lane&15 = column N; lane holds K = kb..kb+7 (elems 0..7)
//                 and K = kb+16..kb+23 (elems 8..15), kb = k0 + (lane>>4)*8.
// C/D (16x16):    VGPR r -> row = r + (lane>>4)*8, col = lane&15.
// ---------------------------------------------------------------------------
__device__ __forceinline__ v16h load_frag_a(const _Float16* base, int ld,
                                            int row, int kb) {
    const _Float16* p = base + (size_t)row * ld + kb;
    v8h lo = *(const v8h*)(p);
    v8h hi = *(const v8h*)(p + 16);
    return __builtin_shufflevector(lo, hi, 0, 1, 2, 3, 4, 5, 6, 7,
                                           8, 9, 10, 11, 12, 13, 14, 15);
}

__device__ __forceinline__ v16h load_frag_b(const _Float16* base, int ld,
                                            int n, int kb) {
    const _Float16* p = base + (size_t)n * ld + kb;
    v8h lo = *(const v8h*)(p);
    v8h hi = *(const v8h*)(p + 16);
    return __builtin_shufflevector(lo, hi, 0, 1, 2, 3, 4, 5, 6, 7,
                                           8, 9, 10, 11, 12, 13, 14, 15);
}

// ---------------------------------------------------------------------------
// One MLP layer for a 64-row x 512-col tile, 8 waves, each wave 64Mx64N
// (4x4 grid of 16x16 accumulator tiles).
// A input = [A0 (K0 cols, global) | A1 (K1 cols, LDS)], weights W are
// [512][ldw] f16 with the same packed K layout. Output (bias + ELU, f16)
// is written to hs[64][512].
// PRESYNC: barrier between finishing reads of hs and overwriting it.
// ---------------------------------------------------------------------------
template <int K0, int K1, bool PRESYNC>
__device__ __forceinline__ void mlp_layer(const _Float16* A0, int ld0,
                                          const _Float16* A1, int ld1,
                                          const _Float16* __restrict__ W, int ldw,
                                          const float* __restrict__ bias,
                                          _Float16* hs,
                                          int wave, int lane) {
    const int mr  = lane & 15;
    const int kh  = lane >> 4;
    const int n0w = wave * 64;

    v8f acc[4][4];
    const v8f vzero = {0.f, 0.f, 0.f, 0.f, 0.f, 0.f, 0.f, 0.f};
#pragma unroll
    for (int i = 0; i < 4; ++i)
#pragma unroll
        for (int j = 0; j < 4; ++j) acc[i][j] = vzero;

    int kw = 0;
#pragma unroll
    for (int k = 0; k < K0; k += 32) {
        v16h a[4];
#pragma unroll
        for (int mt = 0; mt < 4; ++mt)
            a[mt] = load_frag_a(A0, ld0, mt * 16 + mr, k + kh * 8);
#pragma unroll
        for (int nt = 0; nt < 4; ++nt) {
            v16h b = load_frag_b(W, ldw, n0w + nt * 16 + mr, kw + kh * 8);
#pragma unroll
            for (int mt = 0; mt < 4; ++mt)
                acc[mt][nt] = __builtin_amdgcn_wmma_f32_16x16x32_f16(
                    false, a[mt], false, b, (short)0, acc[mt][nt], false, false);
        }
        kw += 32;
    }
#pragma unroll
    for (int k = 0; k < K1; k += 32) {
        v16h a[4];
#pragma unroll
        for (int mt = 0; mt < 4; ++mt)
            a[mt] = load_frag_a(A1, ld1, mt * 16 + mr, k + kh * 8);
#pragma unroll
        for (int nt = 0; nt < 4; ++nt) {
            v16h b = load_frag_b(W, ldw, n0w + nt * 16 + mr, kw + kh * 8);
#pragma unroll
            for (int mt = 0; mt < 4; ++mt)
                acc[mt][nt] = __builtin_amdgcn_wmma_f32_16x16x32_f16(
                    false, a[mt], false, b, (short)0, acc[mt][nt], false, false);
        }
        kw += 32;
    }

    if (PRESYNC) __syncthreads();   // all waves done reading hs before overwrite

    // bias + ELU epilogue, store f16 into LDS h buffer
#pragma unroll
    for (int mt = 0; mt < 4; ++mt) {
#pragma unroll
        for (int nt = 0; nt < 4; ++nt) {
            const int col = n0w + nt * 16 + mr;
            const float bc = bias[col];
#pragma unroll
            for (int r = 0; r < 8; ++r) {
                const int row = mt * 16 + kh * 8 + r;
                float v = acc[mt][nt][r] + bc;
                v = (v > 0.f) ? v : (__expf(v) - 1.f);
                hs[row * HDIM + col] = (_Float16)v;
            }
        }
    }
}

// ---------------------------------------------------------------------------
// Fused 3-layer MLP + output reduction. One block = 64 rows of M.
// Hidden activations live entirely in LDS (never touch HBM).
// ---------------------------------------------------------------------------
__global__ __launch_bounds__(256) void fused_mlp_kernel(
    const _Float16* __restrict__ Xe,    // [M][160]
    const _Float16* __restrict__ W1p,   // [512][160]
    const _Float16* __restrict__ W2p,   // [512][672]
    const _Float16* __restrict__ W3p,   // [512][672]
    const float* __restrict__ b1,
    const float* __restrict__ b2,
    const float* __restrict__ b3,
    const float* __restrict__ Wout,     // [28*512] f32
    float* __restrict__ out)            // [BATCH], pre-initialized to bout
{
    __shared__ _Float16 hs[TM * HDIM];    // 64 KB hidden tile (h1 -> h2 -> h3)

    const int m0   = blockIdx.x * TM;
    const int tid  = threadIdx.x;
    const int wave = tid >> 5;
    const int lane = tid & 31;

    const _Float16* Xt = Xe + (size_t)m0 * K1PAD;   // this block's input rows

    // layer 1: [inp,res](160) @ W1 -> hs
    mlp_layer<K1PAD, 0, false>(Xt, K1PAD, nullptr, 0, W1p, K1PAD, b1, hs, wave, lane);
    __syncthreads();
    // layer 2: [inp(160 packed), h1(512)] @ W2 -> hs
    mlp_layer<K1PAD, HDIM, true>(Xt, K1PAD, hs, HDIM, W2p, K23PAD, b2, hs, wave, lane);
    __syncthreads();
    // layer 3
    mlp_layer<K1PAD, HDIM, true>(Xt, K1PAD, hs, HDIM, W3p, K23PAD, b3, hs, wave, lane);
    __syncthreads();

    // output: out[b] += sum_k h3[row][k] * Wout[(row%28)*512 + k]
    // 4 threads per row, 128 cols each, shuffle-reduce, one atomic per row.
    const int r    = tid >> 2;       // 0..63
    const int lsub = tid & 3;
    const int grow = m0 + r;
    const float* wrow = Wout + (size_t)(grow % NJ) * HDIM;
    float p = 0.f;
#pragma unroll 8
    for (int i = 0; i < 128; ++i) {
        const int c = lsub * 128 + i;
        p += (float)hs[r * HDIM + c] * wrow[c];
    }
    p += __shfl_xor(p, 1);
    p += __shfl_xor(p, 2);
    if (lsub == 0) atomicAdd(&out[grow / NJ], p);
}

// ---------------------------------------------------------------------------
// Feature kernel: geometric residuals + f16 packing of [x, a, r1, r2, pad]
// ---------------------------------------------------------------------------
__device__ __forceinline__ void m6d_rot(const float* d, float R[9]) {
    float a1x = d[0], a1y = d[1], a1z = d[2];
    float a2x = d[3], a2y = d[4], a2z = d[5];
    float n1 = rsqrtf(a1x * a1x + a1y * a1y + a1z * a1z);
    float b1x = a1x * n1, b1y = a1y * n1, b1z = a1z * n1;
    float dt = b1x * a2x + b1y * a2y + b1z * a2z;
    float u2x = a2x - dt * b1x, u2y = a2y - dt * b1y, u2z = a2z - dt * b1z;
    float n2 = rsqrtf(u2x * u2x + u2y * u2y + u2z * u2z);
    float b2x = u2x * n2, b2y = u2y * n2, b2z = u2z * n2;
    float b3x = b1y * b2z - b1z * b2y;
    float b3y = b1z * b2x - b1x * b2z;
    float b3z = b1x * b2y - b1y * b2x;
    R[0] = b1x; R[1] = b1y; R[2] = b1z;
    R[3] = b2x; R[4] = b2y; R[5] = b2z;
    R[6] = b3x; R[7] = b3y; R[8] = b3z;
}

__global__ void feature_kernel(const float* __restrict__ x,
                               const float* __restrict__ a,
                               _Float16* __restrict__ Xe) {
    const int row = blockIdx.x * 256 + threadIdx.x;
    if (row >= MROWS) return;
    const float* xr = x + (size_t)row * XF;
    const float* ar = a + (size_t)row * AF;
    _Float16* o = Xe + (size_t)row * K1PAD;

#pragma unroll 2
    for (int i = 0; i < XF; ++i) o[i] = (_Float16)xr[i];
#pragma unroll
    for (int i = 0; i < AF; ++i) o[XF + i] = (_Float16)ar[i];

    float dx = xr[0] - xr[9], dy = xr[1] - xr[10], dz = xr[2] - xr[11];
    float r1 = sqrtf(dx * dx + dy * dy + dz * dz);

    float R1[9], R2[9];
    m6d_rot(xr + 3, R1);
    m6d_rot(xr + 12, R2);
    float tr = 0.f;
#pragma unroll
    for (int i = 0; i < 9; ++i) tr += R1[i] * R2[i];
    float c = (tr - 1.f) * 0.5f;
    c = fminf(1.f, fmaxf(-1.f, c));
    float ang = acosf(c);

    o[135] = (_Float16)logf(8.f * r1 + 1.f);
    o[136] = (_Float16)logf(80.f * ang + 1.f);
#pragma unroll
    for (int i = 137; i < K1PAD; ++i) o[i] = (_Float16)0.f;
}

// ---------------------------------------------------------------------------
// Weight prep: f32 -> f16 with zero-padded K layout matching the GEMM packing
// ---------------------------------------------------------------------------
__global__ void prep_w1_kernel(const float* __restrict__ W1,
                               _Float16* __restrict__ W1p) {
    const int i = blockIdx.x * 256 + threadIdx.x;
    if (i >= HDIM * K1PAD) return;
    const int n = i / K1PAD, k = i % K1PAD;
    W1p[i] = (k < XF + AF + 2) ? (_Float16)W1[n * (XF + AF + 2) + k]
                               : (_Float16)0.f;
}

__global__ void prep_w23_kernel(const float* __restrict__ W,
                                _Float16* __restrict__ Wp) {
    const int i = blockIdx.x * 256 + threadIdx.x;
    if (i >= HDIM * K23PAD) return;
    const int n = i / K23PAD, k = i % K23PAD;
    float v = 0.f;
    const int IN = XF + AF;                 // 135
    if (k < IN)          v = W[n * (IN + HDIM) + k];                  // inp part
    else if (k >= K1PAD) v = W[n * (IN + HDIM) + (k - (K1PAD - IN))]; // h part
    Wp[i] = (_Float16)v;
}

__global__ void init_out_kernel(const float* __restrict__ bout,
                                float* __restrict__ out) {
    const int i = blockIdx.x * 256 + threadIdx.x;
    if (i < BATCH) out[i] = bout[0];
}

// ---------------------------------------------------------------------------
// Launch
// ---------------------------------------------------------------------------
extern "C" void kernel_launch(void* const* d_in, const int* in_sizes, int n_in,
                              void* d_out, int out_size, void* d_ws, size_t ws_size,
                              hipStream_t stream) {
    (void)in_sizes; (void)n_in; (void)out_size; (void)ws_size;
    const float* x    = (const float*)d_in[0];
    const float* a    = (const float*)d_in[1];
    const float* W1   = (const float*)d_in[2];
    const float* b1   = (const float*)d_in[3];
    const float* W2   = (const float*)d_in[4];
    const float* b2   = (const float*)d_in[5];
    const float* W3   = (const float*)d_in[6];
    const float* b3   = (const float*)d_in[7];
    const float* Wout = (const float*)d_in[8];
    const float* bout = (const float*)d_in[9];
    float* out = (float*)d_out;

    uintptr_t p = (uintptr_t)d_ws;
    auto carve = [&](size_t bytes) {
        void* r = (void*)p;
        p += (bytes + 255) & ~(size_t)255;
        return r;
    };
    _Float16* Xe  = (_Float16*)carve((size_t)MROWS * K1PAD * sizeof(_Float16));
    _Float16* W1p = (_Float16*)carve((size_t)HDIM * K1PAD * sizeof(_Float16));
    _Float16* W2p = (_Float16*)carve((size_t)HDIM * K23PAD * sizeof(_Float16));
    _Float16* W3p = (_Float16*)carve((size_t)HDIM * K23PAD * sizeof(_Float16));

    feature_kernel<<<(MROWS + 255) / 256, 256, 0, stream>>>(x, a, Xe);
    prep_w1_kernel<<<(HDIM * K1PAD + 255) / 256, 256, 0, stream>>>(W1, W1p);
    prep_w23_kernel<<<(HDIM * K23PAD + 255) / 256, 256, 0, stream>>>(W2, W2p);
    prep_w23_kernel<<<(HDIM * K23PAD + 255) / 256, 256, 0, stream>>>(W3, W3p);
    init_out_kernel<<<(BATCH + 255) / 256, 256, 0, stream>>>(bout, out);

    fused_mlp_kernel<<<MROWS / TM, 256, 0, stream>>>(
        Xe, W1p, W2p, W3p, b1, b2, b3, Wout, out);
}
